// MEGNETLayer_46102178955279
// MI455X (gfx1250) — compile-verified, matrix operand order checked by reading, not run on hardware
//
#include <hip/hip_runtime.h>

// ---------------- problem constants (from reference) ----------------
#define N_NODES 50000
#define N_EDGES 800000
#define N_GRAPH 500
#define DIM     128
#define NPG     (N_NODES / N_GRAPH)   // 100 nodes per graph (contiguous)
#define EPG     (N_EDGES / N_GRAPH)   // 1600 edges per graph (contiguous)

typedef __attribute__((ext_vector_type(16))) _Float16 v16h;
typedef __attribute__((ext_vector_type(8)))  _Float16 v8h;
typedef __attribute__((ext_vector_type(8)))  float    v8f;
typedef int v4i_raw __attribute__((vector_size(16)));   // builtin's pointee type

// gfx1250 async memory->LDS path (guarded: fall back to VGPR staging if absent)
#if defined(__has_builtin)
# if __has_builtin(__builtin_amdgcn_global_load_async_to_lds_b128) && \
     __has_builtin(__builtin_amdgcn_s_wait_asynccnt)
#  define HAVE_ASYNC 1
# endif
#endif
#ifndef HAVE_ASYNC
# define HAVE_ASYNC 0
#endif

enum AMode { PLAIN_F32 = 0, PLAIN_F16 = 1, EDGE_CAT = 2, NODE_CAT = 3, STATE_CAT = 4 };

struct GArgs {
  const void* A0; const void* A1; const void* A2;   // mode-dependent A sources
  const int*  src; const int* dst; const int* n2g;
  const _Float16* Wt;    // [Nout][K] f16 (pre-transposed weights)
  const float* bias;     // [Nout] f32
  _Float16* outH;        // optional f16 output (feeds next GEMM)
  float*    outF;        // optional f32 output (final results / readout source)
  int M, Nout;
};

__device__ __forceinline__ float sp2f(float x) {
  // softplus(x) - log(2), stable: max(x,0) + log(1+exp(-|x|)) - log2
  float t = __expf(-fabsf(x));
  return fmaxf(x, 0.0f) + __logf(1.0f + t) - 0.693147180559945f;
}

__device__ __forceinline__ v16h combine(v8h lo, v8h hi) {
  v16h r;
  #pragma unroll
  for (int i = 0; i < 8; ++i) { r[i] = lo[i]; r[i + 8] = hi[i]; }
  return r;
}

// ---------------- templated fused WMMA GEMM (double-buffered) ----------------
// Block = 256 threads = 8 wave32s arranged 2(M) x 4(N); each wave computes a
// 32x32 tile with 4 f32 accumulators -> block tile 64(M) x 128(N), BK = 32.
// Ping-pong LDS: tile t+1 is staged (async memory->LDS when available) while
// tile t is consumed, one barrier per k-step.
#define BM 64
#define BN 128
#define AST 40   // padded LDS stride (halves): 80B rows, conflict-free b128 reads
template <int MODE, int K>
__global__ __launch_bounds__(256) void wmma_gemm(GArgs ga) {
  constexpr int NSTEP = K / 32;
  __shared__ __attribute__((aligned(16))) _Float16 As[2][BM * AST];
  __shared__ __attribute__((aligned(16))) _Float16 Bs[2][BN * AST];

  const int tid  = threadIdx.x;
  const int lane = tid & 31;
  const int wave = tid >> 5;
  const int wm   = wave & 1;            // 0..1 : 32-row sub-tile
  const int wn   = wave >> 1;           // 0..3 : 32-col sub-tile
  const int mBase = blockIdx.x * BM;
  const int nBase = blockIdx.y * BN;

  // ---- per-thread staging role: one v8h chunk of the A tile ----
  const int arow = tid >> 2;            // 0..63
  const int akk  = (tid & 3) * 8;       // 0,8,16,24
  const int grow = mBase + arow;
  const bool aok = grow < ga.M;
  const int rc   = aok ? grow : 0;      // clamped row

  // Hoisted gather indices + per-128-segment row base pointers (computed ONCE).
  const float*    apf[4] = {};
  const _Float16* aph[4] = {};
  if (MODE == PLAIN_F32) {
    const float* b = (const float*)ga.A0 + (size_t)rc * K;
    #pragma unroll
    for (int s = 0; s < K / 128; ++s) apf[s] = b + s * 128;
  } else if (MODE == PLAIN_F16) {
    const _Float16* b = (const _Float16*)ga.A0 + (size_t)rc * K;
    #pragma unroll
    for (int s = 0; s < K / 128; ++s) aph[s] = b + s * 128;
  } else if (MODE == EDGE_CAT) {        // [h1[src], h1[dst], e1, s1[n2g[src]]]
    const int sr = ga.src[rc], dr = ga.dst[rc];
    aph[0] = (const _Float16*)ga.A0 + (size_t)sr * DIM;
    aph[1] = (const _Float16*)ga.A0 + (size_t)dr * DIM;
    aph[2] = (const _Float16*)ga.A1 + (size_t)rc * DIM;
    aph[3] = (const _Float16*)ga.A2 + (size_t)ga.n2g[sr] * DIM;
  } else if (MODE == NODE_CAT) {        // [hh, h1, s1[n2g]]
    aph[0] = (const _Float16*)ga.A0 + (size_t)rc * DIM;
    aph[1] = (const _Float16*)ga.A1 + (size_t)rc * DIM;
    aph[2] = (const _Float16*)ga.A2 + (size_t)ga.n2g[rc] * DIM;
  } else {                              // STATE_CAT: [s1, u_edge, u_node]
    aph[0] = (const _Float16*)ga.A0 + (size_t)rc * DIM;
    aph[1] = (const _Float16*)ga.A1 + (size_t)rc * DIM;
    aph[2] = (const _Float16*)ga.A2 + (size_t)rc * DIM;
  }

  const _Float16* wt = ga.Wt;

  // Stage one 32-wide k-slice of A and B into LDS buffer `buf`.
  auto stage = [&](int buf, int step) {
    const int seg = step >> 2;          // (step*32)/128
    const int off = (step & 3) * 32;    // offset within 128-segment
    _Float16* as = &As[buf][0];
    _Float16* bs = &Bs[buf][0];
    // ---- A tile: one b128 per thread (gathered row slice) ----
    if (MODE == PLAIN_F32) {
      v8h av = {};
      if (aok) {
        const float* p = apf[seg] + off + akk;
        float4 f0 = *(const float4*)p;
        float4 f1 = *(const float4*)(p + 4);
        av[0] = (_Float16)f0.x; av[1] = (_Float16)f0.y;
        av[2] = (_Float16)f0.z; av[3] = (_Float16)f0.w;
        av[4] = (_Float16)f1.x; av[5] = (_Float16)f1.y;
        av[6] = (_Float16)f1.z; av[7] = (_Float16)f1.w;
      }
      *(v8h*)&as[arow * AST + akk] = av;
    } else {
#if HAVE_ASYNC
      if (aok) {
        __builtin_amdgcn_global_load_async_to_lds_b128(
            (v4i_raw*)(aph[seg] + off + akk),
            (v4i_raw*)&as[arow * AST + akk], 0, 0);
      } else {
        v8h z = {};
        *(v8h*)&as[arow * AST + akk] = z;
      }
#else
      v8h av = {};
      if (aok) av = *(const v8h*)(aph[seg] + off + akk);
      *(v8h*)&as[arow * AST + akk] = av;
#endif
    }
    // ---- B tile: two b128 per thread from f16 [Nout][K] weights ----
    #pragma unroll
    for (int i = 0; i < 2; ++i) {
      int idx = i * 256 + tid;
      int nn = idx >> 2, kk8 = (idx & 3) * 8;
      const _Float16* gp = wt + (size_t)(nBase + nn) * K + step * 32 + kk8;
#if HAVE_ASYNC
      __builtin_amdgcn_global_load_async_to_lds_b128(
          (v4i_raw*)gp, (v4i_raw*)&bs[nn * AST + kk8], 0, 0);
#else
      *(v8h*)&bs[nn * AST + kk8] = *(const v8h*)gp;
#endif
    }
  };

  auto tile_sync = [&]() {
#if HAVE_ASYNC
    __builtin_amdgcn_s_wait_asynccnt(0);
#endif
    __syncthreads();
  };

  v8f acc00 = {}, acc01 = {}, acc10 = {}, acc11 = {};
  const int lr = lane & 15;
  const int kb = (lane >> 4) * 8;       // 0 or 8 per documented 16-bit layout

  stage(0, 0);                          // prologue
  #pragma unroll
  for (int step = 0; step < NSTEP; ++step) {
    const int cur = step & 1;
    tile_sync();                        // tile[cur] ready in LDS
    if (step + 1 < NSTEP) stage(cur ^ 1, step + 1);  // overlap next stage
    // ---- fragments (ds_load_b128 pairs) + 4 WMMAs from tile[cur] ----
    const _Float16* ar0 = &As[cur][(wm * 32 + lr) * AST];
    const _Float16* ar1 = &As[cur][(wm * 32 + 16 + lr) * AST];
    const _Float16* br0 = &Bs[cur][(wn * 32 + lr) * AST];
    const _Float16* br1 = &Bs[cur][(wn * 32 + 16 + lr) * AST];
    v16h a0 = combine(*(const v8h*)(ar0 + kb), *(const v8h*)(ar0 + kb + 16));
    v16h a1 = combine(*(const v8h*)(ar1 + kb), *(const v8h*)(ar1 + kb + 16));
    v16h b0 = combine(*(const v8h*)(br0 + kb), *(const v8h*)(br0 + kb + 16));
    v16h b1 = combine(*(const v8h*)(br1 + kb), *(const v8h*)(br1 + kb + 16));
    acc00 = __builtin_amdgcn_wmma_f32_16x16x32_f16(false, a0, false, b0, (short)0, acc00, false, false);
    acc01 = __builtin_amdgcn_wmma_f32_16x16x32_f16(false, a0, false, b1, (short)0, acc01, false, false);
    acc10 = __builtin_amdgcn_wmma_f32_16x16x32_f16(false, a1, false, b0, (short)0, acc10, false, false);
    acc11 = __builtin_amdgcn_wmma_f32_16x16x32_f16(false, a1, false, b1, (short)0, acc11, false, false);
  }

  // ---- epilogue: bias + SoftPlus2, dual f32/f16 writeback ----
  // C/D layout: lanes 0-15 -> N=lane, M=r; lanes 16-31 -> N=lane-16, M=8+r.
  const int nb = nBase + wn * 32 + (lane & 15);
  const int mb = mBase + wm * 32 + (lane >> 4) * 8;
  const float bv0 = ga.bias[nb];
  const float bv1 = ga.bias[nb + 16];
  #pragma unroll
  for (int r = 0; r < 8; ++r) {
    int m0 = mb + r, m1 = mb + 16 + r;
    if (m0 < ga.M) {
      float v0 = sp2f(acc00[r] + bv0);
      float v1 = sp2f(acc01[r] + bv1);
      size_t o = (size_t)m0 * ga.Nout + nb;
      if (ga.outF) { ga.outF[o] = v0; ga.outF[o + 16] = v1; }
      if (ga.outH) { ga.outH[o] = (_Float16)v0; ga.outH[o + 16] = (_Float16)v1; }
    }
    if (m1 < ga.M) {
      float v0 = sp2f(acc10[r] + bv0);
      float v1 = sp2f(acc11[r] + bv1);
      size_t o = (size_t)m1 * ga.Nout + nb;
      if (ga.outF) { ga.outF[o] = v0; ga.outF[o + 16] = v1; }
      if (ga.outH) { ga.outH[o] = (_Float16)v0; ga.outH[o + 16] = (_Float16)v1; }
    }
  }
}

// ---------------- weight transpose + f16 convert: W[K][N] -> Wt[N][K] ----------------
__global__ __launch_bounds__(256) void wtrans_kernel(const float* __restrict__ W,
                                                     _Float16* __restrict__ Wt,
                                                     int K, int Nout) {
  int i = blockIdx.x * 256 + threadIdx.x;
  if (i < K * Nout) {
    int k = i / Nout, n = i % Nout;
    Wt[(size_t)n * K + k] = (_Float16)W[i];
  }
}

// ---------------- segment mean over dst (per-graph, LDS atomics) ----------------
__global__ __launch_bounds__(256) void hh_mean_kernel(const float* __restrict__ e2,
                                                      const int* __restrict__ dst,
                                                      _Float16* __restrict__ hh_h) {
  __shared__ float acc[NPG * DIM];   // 51.2 KB of the 320 KB LDS
  __shared__ float cnt[NPG];
  const int g = blockIdx.x, tid = threadIdx.x;
  for (int i = tid; i < NPG * DIM; i += 256) acc[i] = 0.0f;
  for (int i = tid; i < NPG; i += 256) cnt[i] = 0.0f;
  __syncthreads();
  const int e0 = g * EPG;
  for (int i = tid; i < EPG; i += 256)
    atomicAdd(&cnt[dst[e0 + i] - g * NPG], 1.0f);
  for (int i = tid; i < EPG * DIM; i += 256) {
    int el = i >> 7, c = i & 127;
    int ln = dst[e0 + el] - g * NPG;
    atomicAdd(&acc[ln * DIM + c], e2[(size_t)(e0 + el) * DIM + c]);
  }
  __syncthreads();
  for (int i = tid; i < NPG * DIM; i += 256) {
    int ln = i >> 7;
    float cf = fmaxf(cnt[ln], 1.0f);
    hh_h[(size_t)(g * NPG + ln) * DIM + (i & 127)] = (_Float16)(acc[i] / cf);
  }
}

// ---------------- per-graph mean readout (contiguous rows, no atomics) ----------------
__global__ __launch_bounds__(128) void gmean_kernel(const float* __restrict__ x,
                                                    int rows_per_g,
                                                    _Float16* __restrict__ out) {
  const int g = blockIdx.x, c = threadIdx.x;   // 128 threads = one column each
  float s = 0.0f;
  size_t base = (size_t)g * rows_per_g * DIM;
  for (int r = 0; r < rows_per_g; ++r) s += x[base + (size_t)r * DIM + c];
  out[(size_t)g * DIM + c] = (_Float16)(s / (float)rows_per_g);
}

// ---------------- host side ----------------
template <int MODE, int K>
static void run_gemm(hipStream_t st,
                     const void* A0, const void* A1, const void* A2,
                     const int* src, const int* dst, const int* n2g,
                     const _Float16* Wt, const float* bias,
                     _Float16* outH, float* outF, int M, int Nout) {
  GArgs ga{A0, A1, A2, src, dst, n2g, Wt, bias, outH, outF, M, Nout};
  dim3 grid((M + BM - 1) / BM, Nout / BN);
  hipLaunchKernelGGL((wmma_gemm<MODE, K>), grid, dim3(256), 0, st, ga);
}

extern "C" void kernel_launch(void* const* d_in, const int* in_sizes, int n_in,
                              void* d_out, int out_size, void* d_ws, size_t ws_size,
                              hipStream_t stream) {
  (void)in_sizes; (void)n_in; (void)out_size; (void)ws_size;
  // inputs in setup_inputs() dict order
  const float* h = (const float*)d_in[0];
  const float* e = (const float*)d_in[1];
  const float* s = (const float*)d_in[2];
  const int* src = (const int*)d_in[33];
  const int* dst = (const int*)d_in[34];
  const int* n2g = (const int*)d_in[35];

  // outputs: (h2[N,D], e2[E,D], s2[G,D]) concatenated, fp32
  float* h2_f = (float*)d_out;
  float* e2_f = h2_f + (size_t)N_NODES * DIM;
  float* s2_f = e2_f + (size_t)N_EDGES * DIM;

  // workspace carve-up
  char* ws = (char*)d_ws;
  size_t off = 0;
  auto carve = [&](size_t bytes) { char* p = ws + off; off += (bytes + 255) & ~size_t(255); return p; };
  _Float16* h1h  = (_Float16*)carve((size_t)N_NODES * DIM * 2);
  _Float16* e1h  = (_Float16*)carve((size_t)N_EDGES * DIM * 2);
  _Float16* s1h  = (_Float16*)carve((size_t)N_GRAPH * DIM * 2);
  _Float16* hidA = (_Float16*)carve((size_t)N_EDGES * 256 * 2);
  _Float16* hidB = (_Float16*)carve((size_t)N_EDGES * 256 * 2);
  _Float16* hh_h = (_Float16*)carve((size_t)N_NODES * DIM * 2);
  _Float16* ue_h = (_Float16*)carve((size_t)N_GRAPH * DIM * 2);
  _Float16* un_h = (_Float16*)carve((size_t)N_GRAPH * DIM * 2);

  // pre-transpose all weight matrices to f16 [Nout][K] (≈2 MB, L2-resident)
  struct WD { int idx, K, N; _Float16* Wt; };
  WD wd[16] = {
    {3,  128, 256, nullptr}, {5,  256, 128, nullptr},   // node_func
    {7,  128, 256, nullptr}, {9,  256, 128, nullptr},   // edge_func
    {11, 128, 256, nullptr}, {13, 256, 128, nullptr},   // state_func
    {15, 512, 256, nullptr}, {17, 256, 256, nullptr}, {19, 256, 128, nullptr},  // edge_update
    {21, 384, 256, nullptr}, {23, 256, 256, nullptr}, {25, 256, 128, nullptr},  // node_update
    {27, 384, 256, nullptr}, {29, 256, 256, nullptr}, {31, 256, 128, nullptr},  // state_update
    {0, 0, 0, nullptr}
  };
  for (int i = 0; i < 15; ++i) {
    wd[i].Wt = (_Float16*)carve((size_t)wd[i].K * wd[i].N * 2);
    int tot = wd[i].K * wd[i].N;
    hipLaunchKernelGGL(wtrans_kernel, dim3((tot + 255) / 256), dim3(256), 0, stream,
                       (const float*)d_in[wd[i].idx], wd[i].Wt, wd[i].K, wd[i].N);
  }
  const _Float16 *nfT1 = wd[0].Wt,  *nfT2 = wd[1].Wt;
  const _Float16 *efT1 = wd[2].Wt,  *efT2 = wd[3].Wt;
  const _Float16 *sfT1 = wd[4].Wt,  *sfT2 = wd[5].Wt;
  const _Float16 *euT1 = wd[6].Wt,  *euT2 = wd[7].Wt,  *euT3 = wd[8].Wt;
  const _Float16 *nuT1 = wd[9].Wt,  *nuT2 = wd[10].Wt, *nuT3 = wd[11].Wt;
  const _Float16 *suT1 = wd[12].Wt, *suT2 = wd[13].Wt, *suT3 = wd[14].Wt;
  const float *nfb1 = (const float*)d_in[4],  *nfb2 = (const float*)d_in[6];
  const float *efb1 = (const float*)d_in[8],  *efb2 = (const float*)d_in[10];
  const float *sfb1 = (const float*)d_in[12], *sfb2 = (const float*)d_in[14];
  const float *eub1 = (const float*)d_in[16], *eub2 = (const float*)d_in[18], *eub3 = (const float*)d_in[20];
  const float *nub1 = (const float*)d_in[22], *nub2 = (const float*)d_in[24], *nub3 = (const float*)d_in[26];
  const float *sub1 = (const float*)d_in[28], *sub2 = (const float*)d_in[30], *sub3 = (const float*)d_in[32];

  // 1) pre-update MLPs (sp2 after every layer)
  run_gemm<PLAIN_F32, 128>(stream, h, 0, 0, 0, 0, 0, nfT1, nfb1, hidA, nullptr, N_NODES, 256);
  run_gemm<PLAIN_F16, 256>(stream, hidA, 0, 0, 0, 0, 0, nfT2, nfb2, h1h, nullptr, N_NODES, 128);
  run_gemm<PLAIN_F32, 128>(stream, e, 0, 0, 0, 0, 0, efT1, efb1, hidA, nullptr, N_EDGES, 256);
  run_gemm<PLAIN_F16, 256>(stream, hidA, 0, 0, 0, 0, 0, efT2, efb2, e1h, nullptr, N_EDGES, 128);
  run_gemm<PLAIN_F32, 128>(stream, s, 0, 0, 0, 0, 0, sfT1, sfb1, hidA, nullptr, N_GRAPH, 256);
  run_gemm<PLAIN_F16, 256>(stream, hidA, 0, 0, 0, 0, 0, sfT2, sfb2, s1h, nullptr, N_GRAPH, 128);

  // 2) edge update: concat fused into A staging (no [E,512] buffer ever exists)
  run_gemm<EDGE_CAT, 512>(stream, h1h, e1h, s1h, src, dst, n2g, euT1, eub1, hidA, nullptr, N_EDGES, 256);
  run_gemm<PLAIN_F16, 256>(stream, hidA, 0, 0, 0, 0, 0, euT2, eub2, hidB, nullptr, N_EDGES, 256);
  run_gemm<PLAIN_F16, 256>(stream, hidB, 0, 0, 0, 0, 0, euT3, eub3, nullptr, e2_f, N_EDGES, 128);

  // 3) segment mean over dst (per-graph contiguous ranges, LDS atomics)
  hipLaunchKernelGGL(hh_mean_kernel, dim3(N_GRAPH), dim3(256), 0, stream, e2_f, dst, hh_h);

  // 4) node update
  run_gemm<NODE_CAT, 384>(stream, hh_h, h1h, s1h, 0, 0, n2g, nuT1, nub1, hidA, nullptr, N_NODES, 256);
  run_gemm<PLAIN_F16, 256>(stream, hidA, 0, 0, 0, 0, 0, nuT2, nub2, hidB, nullptr, N_NODES, 256);
  run_gemm<PLAIN_F16, 256>(stream, hidB, 0, 0, 0, 0, 0, nuT3, nub3, nullptr, h2_f, N_NODES, 128);

  // 5) per-graph mean readouts (contiguous rows, no atomics)
  hipLaunchKernelGGL(gmean_kernel, dim3(N_GRAPH), dim3(128), 0, stream, e2_f, EPG, ue_h);
  hipLaunchKernelGGL(gmean_kernel, dim3(N_GRAPH), dim3(128), 0, stream, h2_f, NPG, un_h);

  // 6) state update
  run_gemm<STATE_CAT, 384>(stream, s1h, ue_h, un_h, 0, 0, 0, suT1, sub1, hidA, nullptr, N_GRAPH, 256);
  run_gemm<PLAIN_F16, 256>(stream, hidA, 0, 0, 0, 0, 0, suT2, sub2, hidB, nullptr, N_GRAPH, 256);
  run_gemm<PLAIN_F16, 256>(stream, hidB, 0, 0, 0, 0, 0, suT3, sub3, nullptr, s2_f, N_GRAPH, 128);
}
